// Attention_10307921511049
// MI455X (gfx1250) — matrix operand order repaired and checked by reading, not compiled, not run on hardware
//
#include <hip/hip_runtime.h>
#include <hip/hip_bf16.h>
#include <math.h>

// ---------------------------------------------------------------------------
// Shapes (compile-time constants from the reference)
// ---------------------------------------------------------------------------
#define B_  2
#define N_  2048
#define D_  1024
#define H_  16
#define DH_ 64
#define C_  64
#define SCALE_ 8.0f
#define L_   (N_ + 1)     // 2049 kv rows (null + N)
#define LP_  2080         // kv rows padded to multiple of 32
#define BIAS_LD_ (C_ + N_) // 2112

typedef __bf16 bf16_t;
typedef __attribute__((ext_vector_type(16))) __bf16 bf16x16;
typedef __attribute__((ext_vector_type(8)))  __bf16 bf16x8;
typedef __attribute__((ext_vector_type(8)))  float  f32x8;
typedef __attribute__((ext_vector_type(4)))  unsigned int u32x4;
typedef __attribute__((ext_vector_type(8)))  unsigned int u32x8;

static __device__ __forceinline__ f32x8 wmma_bf16(bf16x16 a, bf16x16 b, f32x8 c) {
    return __builtin_amdgcn_wmma_f32_16x16x32_bf16(false, a, false, b, (short)0, c,
                                                   false, false);
}

static __device__ __forceinline__ f32x8 f32x8_zero() {
    f32x8 z;
#pragma unroll
    for (int i = 0; i < 8; ++i) z[i] = 0.0f;
    return z;
}

// ---------------------------------------------------------------------------
// Fragment loaders for V_WMMA_F32_16X16X32_BF16 (wave32).
//
// A (16x32, row-major source, rows = M, contiguous K per row):
//   lane 0-15 : row M=lane,    elems[0..7]=K0..7,   elems[8..15]=K16..23
//   lane16-31 : row M=lane-16, elems[0..7]=K8..15,  elems[8..15]=K24..31
//
// B (32x16; source holds each OUTPUT COLUMN as a contiguous K-run):
//   lane 0-15 : col N=lane,    elems[0..7]=K0..7,   elems[8..15]=K16..23
//   lane16-31 : col N=lane-16, elems[0..7]=K8..15,  elems[8..15]=K24..31
// ---------------------------------------------------------------------------
static __device__ __forceinline__ bf16x16
frag_a(const bf16_t* __restrict__ p, int ld, int row0, int k0) {
    const int lane = threadIdx.x & 31;
    const int r    = row0 + (lane & 15);
    const int hl   = lane >> 4;
    const bf16_t* base = p + (long long)r * ld + k0 + hl * 8;
    bf16x8 lo = *(const bf16x8*)(base);
    bf16x8 hi = *(const bf16x8*)(base + 16);
    bf16x16 o;
#pragma unroll
    for (int i = 0; i < 8; ++i) { o[i] = lo[i]; o[i + 8] = hi[i]; }
    return o;
}

static __device__ __forceinline__ bf16x16
frag_b(const bf16_t* __restrict__ p, int ld, int col0, int k0) {
    const int lane = threadIdx.x & 31;
    const int c    = col0 + (lane & 15);
    const int hl   = lane >> 4;
    const bf16_t* base = p + (long long)c * ld + k0 + hl * 8;
    bf16x8 lo = *(const bf16x8*)(base);
    bf16x8 hi = *(const bf16x8*)(base + 16);
    bf16x16 o;
#pragma unroll
    for (int i = 0; i < 8; ++i) { o[i] = lo[i]; o[i + 8] = hi[i]; }
    return o;
}

// C/D layout (16x16 f32, 8 VGPRs): element (m, n) -> lane n + 16*(m>=8),
// vector slot m%8.  Per lane: n = lane&15, rows m = slot + 8*(lane>>4).

// ---------------------------------------------------------------------------
// Kernel 1: cast + transpose weights to bf16 (Wt[n][k] = W[k][n])
// ---------------------------------------------------------------------------
__global__ __launch_bounds__(256) void prep_weights_kernel(
    const float* __restrict__ Wq, const float* __restrict__ Wkv,
    const float* __restrict__ Wout,
    bf16_t* __restrict__ WqT, bf16_t* __restrict__ WkT,
    bf16_t* __restrict__ WvT, bf16_t* __restrict__ WoutT) {
    int idx = blockIdx.x * 256 + threadIdx.x;     // over 1024*1024
    int n = idx >> 10;
    int k = idx & 1023;
    WqT[idx]   = (bf16_t)Wq[(long long)k * 1024 + n];
    WkT[idx]   = (bf16_t)Wkv[(long long)k * 2048 + n];
    WvT[idx]   = (bf16_t)Wkv[(long long)k * 2048 + 1024 + n];
    WoutT[idx] = (bf16_t)Wout[(long long)k * 1024 + n];
}

// ---------------------------------------------------------------------------
// Kernel 2: LayerNorm -> bf16.  One 256-thread block per (b,n) row of 1024.
// ---------------------------------------------------------------------------
__global__ __launch_bounds__(256) void ln_kernel(
    const float* __restrict__ x, const float* __restrict__ gamma,
    bf16_t* __restrict__ xn) {
    __shared__ float sb[256];
    const int row = blockIdx.x;
    const int tid = threadIdx.x;
    const float* xr = x + (long long)row * D_;

    float v[4];
    float s = 0.f;
#pragma unroll
    for (int i = 0; i < 4; ++i) { v[i] = xr[tid * 4 + i]; s += v[i]; }
    sb[tid] = s; __syncthreads();
    for (int st = 128; st > 0; st >>= 1) {
        if (tid < st) sb[tid] += sb[tid + st];
        __syncthreads();
    }
    const float mu = sb[0] * (1.0f / D_);
    __syncthreads();

    float s2 = 0.f;
#pragma unroll
    for (int i = 0; i < 4; ++i) { float d = v[i] - mu; s2 += d * d; }
    sb[tid] = s2; __syncthreads();
    for (int st = 128; st > 0; st >>= 1) {
        if (tid < st) sb[tid] += sb[tid + st];
        __syncthreads();
    }
    const float rstd = rsqrtf(sb[0] * (1.0f / D_) + 1e-5f);

    bf16_t* xo = xn + (long long)row * D_;
#pragma unroll
    for (int i = 0; i < 4; ++i) {
        int c = tid * 4 + i;
        xo[c] = (bf16_t)((v[i] - mu) * rstd * gamma[c]);
    }
}

// ---------------------------------------------------------------------------
// Kernel 3: null-kv row + zero padding of K / Vt planes.
// ---------------------------------------------------------------------------
__global__ __launch_bounds__(64) void init_kv_kernel(
    const float* __restrict__ null_kv, const float* __restrict__ k_scale,
    bf16_t* __restrict__ K, bf16_t* __restrict__ Vt) {
    const int bh = blockIdx.x;      // 0..B_*H_-1
    const int h  = bh & (H_ - 1);
    const int dh = threadIdx.x;     // 0..63
    __shared__ float red[64];

    const float kn = null_kv[h * DH_ + dh];               // null_kv[0][h][0][dh]
    const float vn = null_kv[H_ * DH_ + h * DH_ + dh];    // null_kv[1][h][0][dh]
    red[dh] = kn * kn; __syncthreads();
    for (int st = 32; st > 0; st >>= 1) {
        if (dh < st) red[dh] += red[dh + st];
        __syncthreads();
    }
    const float inv = 1.0f / fmaxf(sqrtf(red[0]), 1e-12f);

    bf16_t* Kp = K  + (long long)bh * LP_ * DH_;
    bf16_t* Vp = Vt + (long long)bh * DH_ * LP_;
    Kp[dh]                  = (bf16_t)(kn * inv * k_scale[dh]);
    Vp[(long long)dh * LP_] = (bf16_t)vn;
    for (int j = L_; j < LP_; ++j) {
        Kp[(long long)j * DH_ + dh]  = (bf16_t)0.0f;
        Vp[(long long)dh * LP_ + j]  = (bf16_t)0.0f;
    }
}

// ---------------------------------------------------------------------------
// Kernel 4: fused QKV projection (WMMA) + l2norm/scale epilogue.
// One block per (b, h, 16-token tile); 3 waves: q / k / v.
//
// The 16x1024 bf16 xn A-tile (32 KB) is staged into LDS ONCE per block by a
// single Tensor-Data-Mover descriptor (tensor_load_to_lds), issued by wave 0
// and fenced with s_wait_tensorcnt + barrier.  All 3 waves then read their
// A-fragments from LDS, eliminating the 3x redundant global A traffic.
// ---------------------------------------------------------------------------
__global__ __launch_bounds__(96) void qkv_kernel(
    const bf16_t* __restrict__ xn,
    const bf16_t* __restrict__ WqT, const bf16_t* __restrict__ WkT,
    const bf16_t* __restrict__ WvT,
    const float* __restrict__ q_scale, const float* __restrict__ k_scale,
    bf16_t* __restrict__ Q, bf16_t* __restrict__ K, bf16_t* __restrict__ Vt) {
    const int tile = blockIdx.x & 127;
    const int h    = (blockIdx.x >> 7) & (H_ - 1);
    const int b    = blockIdx.x >> 11;
    const int wave = threadIdx.x >> 5;
    const int lane = threadIdx.x & 31;
    const int i0   = tile * 16;

    const bf16_t* WT = (wave == 0) ? WqT : (wave == 1) ? WkT : WvT;

    __shared__ __align__(16) bf16_t xtile[16][D_];   // 32 KB staged A tile
    __shared__ float ts[3][16][65];

    // --- TDM: stage xn[b, i0..i0+15, 0..1023] (bf16) into LDS --------------
    if (wave == 0) {
        const unsigned long long gaddr =
            (unsigned long long)(uintptr_t)(xn + ((long long)b * N_ + i0) * D_);
        // LDS aperture keeps the byte offset in addr[31:0] of the flat pointer.
        const unsigned lds_off = __builtin_amdgcn_readfirstlane(
            (unsigned)(uintptr_t)&xtile[0][0]);
        const unsigned g_lo = __builtin_amdgcn_readfirstlane((unsigned)gaddr);
        const unsigned g_hi = __builtin_amdgcn_readfirstlane(
            (unsigned)(gaddr >> 32));

        u32x4 g0;
        g0[0] = 1u;                                   // count=1, user descriptor
        g0[1] = lds_off;                              // lds_addr
        g0[2] = g_lo;                                 // global_addr[31:0]
        g0[3] = (g_hi & 0x01FFFFFFu) | 0x80000000u;   // addr[56:32] | type=2

        u32x8 g1;
        g1[0] = 0x00010000u;   // workgroup_mask=0, data_size=1 (2 bytes)
        g1[1] = (unsigned)(D_) << 16;      // tensor_dim0[15:0]=1024 @ bits 79:48
        g1[2] = 16u << 16;                 // tensor_dim0 hi=0 | tensor_dim1=16
        g1[3] = (unsigned)(D_) << 16;      // tensor_dim1 hi=0 | tile_dim0=1024
        g1[4] = 16u;                       // tile_dim1=16, tile_dim2=0
        g1[5] = (unsigned)(D_);            // tensor_dim0_stride[31:0]=1024
        g1[6] = 0u;                        // stride hi, tensor_dim1_stride lo
        g1[7] = 0u;

        asm volatile("tensor_load_to_lds %0, %1" :: "s"(g0), "s"(g1) : "memory");
        __builtin_amdgcn_s_wait_tensorcnt(0);
    }
    __syncthreads();

    f32x8 acc[4];
#pragma unroll
    for (int i = 0; i < 4; ++i) acc[i] = f32x8_zero();

    for (int k0 = 0; k0 < D_; k0 += 32) {
        bf16x16 a = frag_a(&xtile[0][0], D_, 0, k0);   // ds_load_b128 path
#pragma unroll
        for (int nt = 0; nt < 4; ++nt) {
            bf16x16 bb = frag_b(WT, D_, h * DH_ + nt * 16, k0);
            acc[nt] = wmma_bf16(a, bb, acc[nt]);
        }
    }

    // spill C-layout tiles to LDS, then per-row epilogues
    const int half = lane >> 4, nn = lane & 15;
#pragma unroll
    for (int nt = 0; nt < 4; ++nt)
#pragma unroll
        for (int v = 0; v < 8; ++v)
            ts[wave][v + 8 * half][nt * 16 + nn] = acc[nt][v];
    __syncthreads();

    if (lane < 16) {
        const int r = lane;
        const long long plane = (long long)(b * H_ + h);
        if (wave == 0) {                               // q: l2norm * q_scale
            float ss = 0.f;
            for (int d = 0; d < DH_; ++d) { float t = ts[0][r][d]; ss += t * t; }
            const float inv = 1.0f / fmaxf(sqrtf(ss), 1e-12f);
            bf16_t* qr = Q + (plane * N_ + i0 + r) * DH_;
            for (int d = 0; d < DH_; ++d)
                qr[d] = (bf16_t)(ts[0][r][d] * inv * q_scale[d]);
        } else if (wave == 1) {                        // k: l2norm * k_scale
            float ss = 0.f;
            for (int d = 0; d < DH_; ++d) { float t = ts[1][r][d]; ss += t * t; }
            const float inv = 1.0f / fmaxf(sqrtf(ss), 1e-12f);
            bf16_t* kr = K + (plane * LP_ + 1 + i0 + r) * DH_;
            for (int d = 0; d < DH_; ++d)
                kr[d] = (bf16_t)(ts[1][r][d] * inv * k_scale[d]);
        } else {                                       // v: transposed store
            bf16_t* vp = Vt + plane * DH_ * LP_;
            for (int d = 0; d < DH_; ++d)
                vp[(long long)d * LP_ + 1 + i0 + r] = (bf16_t)ts[2][r][d];
        }
    }
}

// ---------------------------------------------------------------------------
// Kernel 5: flash attention.  One wave per (b, h, 16-query tile).
// j-tiles of 32: S = Q(16x64) Kt(64x32) -> 4 WMMAs; P@V -> 4 WMMAs.
// ---------------------------------------------------------------------------
__global__ __launch_bounds__(32) void attn_kernel(
    const bf16_t* __restrict__ Q, const bf16_t* __restrict__ K,
    const bf16_t* __restrict__ Vt,
    const float* __restrict__ attn_bias, const unsigned char* __restrict__ cmask,
    bf16_t* __restrict__ O) {
    const int tile = blockIdx.x & 127;
    const int h    = (blockIdx.x >> 7) & (H_ - 1);
    const int b    = blockIdx.x >> 11;
    const int lane = threadIdx.x & 31;
    const int half = lane >> 4, nn = lane & 15;
    const int i0   = tile * 16;

    const long long plane = (long long)(b * H_ + h);
    const bf16_t* Qp = Q  + plane * N_ * DH_;
    const bf16_t* Kp = K  + plane * LP_ * DH_;
    const bf16_t* Vp = Vt + plane * DH_ * LP_;

    bf16x16 aq0 = frag_a(Qp, DH_, i0, 0);
    bf16x16 aq1 = frag_a(Qp, DH_, i0, 32);

    f32x8 o_acc[4];
#pragma unroll
    for (int i = 0; i < 4; ++i) o_acc[i] = f32x8_zero();

    float m_st = -3.0e38f;   // row state, meaningful in lanes 0..15
    float l_st = 0.0f;

    __shared__ float            s_tile[16][33];
    __shared__ __align__(16) bf16_t p_tile[16][32];
    __shared__ float            alpha_s[16];
    __shared__ float            linv_s[16];

    for (int t = 0; t < LP_ / 32; ++t) {
        const int j0 = t * 32;
        if (t + 1 < LP_ / 32) {   // gfx1250 global_prefetch_b8 path
            __builtin_prefetch(Kp + (long long)(j0 + 32 + lane) * DH_, 0, 0);
            __builtin_prefetch(Vp + (long long)(lane & 15) * LP_ + j0 + 32, 0, 0);
        }

        // --- S = (Q . K^T) over DH=64 -------------------------------------
        f32x8 s0 = f32x8_zero(), s1 = f32x8_zero();
#pragma unroll
        for (int kk = 0; kk < 2; ++kk) {
            bf16x16 a  = kk ? aq1 : aq0;
            bf16x16 b0 = frag_b(Kp, DH_, j0,      kk * 32);
            bf16x16 b1 = frag_b(Kp, DH_, j0 + 16, kk * 32);
            s0 = wmma_bf16(a, b0, s0);
            s1 = wmma_bf16(a, b1, s1);
        }

        // --- scale, bias, mask -> LDS (C layout) --------------------------
#pragma unroll
        for (int vh = 0; vh < 2; ++vh) {
            f32x8 sv = vh ? s1 : s0;
            const int j = j0 + vh * 16 + nn;
#pragma unroll
            for (int v = 0; v < 8; ++v) {
                const int mrow = v + 8 * half;
                float val = sv[v] * SCALE_;
                bool valid;
                if (j == 0) {
                    valid = true;
                } else if (j <= N_) {
                    valid = cmask[(long long)b * N_ + (j - 1)] != 0;
                    val += attn_bias[(long long)(C_ + i0 + mrow) * BIAS_LD_ +
                                     C_ + (j - 1)];
                } else {
                    valid = false;     // padded kv rows
                }
                s_tile[mrow][vh * 16 + nn] = valid ? val : -3.0e38f;
            }
        }
        __syncthreads();

        // --- online softmax (lanes 0..15 own one row each) ----------------
        if (lane < 16) {
            const int r = lane;
            float mx = m_st;
            for (int jj = 0; jj < 32; ++jj) mx = fmaxf(mx, s_tile[r][jj]);
            const float al = __expf(m_st - mx);
            float sum = 0.f;
            for (int jj = 0; jj < 32; ++jj) {
                float p = __expf(s_tile[r][jj] - mx);
                sum += p;
                p_tile[r][jj] = (bf16_t)p;
            }
            l_st = l_st * al + sum;
            m_st = mx;
            alpha_s[r] = al;
        }
        __syncthreads();

        // --- rescale O, accumulate P @ V ----------------------------------
        float alv[8];
#pragma unroll
        for (int v = 0; v < 8; ++v) alv[v] = alpha_s[v + 8 * half];
#pragma unroll
        for (int nt = 0; nt < 4; ++nt)
#pragma unroll
            for (int v = 0; v < 8; ++v) o_acc[nt][v] *= alv[v];

        bf16x16 ap = frag_a(&p_tile[0][0], 32, 0, 0);
#pragma unroll
        for (int nt = 0; nt < 4; ++nt) {
            bf16x16 bv = frag_b(Vp, LP_, nt * 16, j0);  // Vt rows = dh columns
            o_acc[nt] = wmma_bf16(ap, bv, o_acc[nt]);
        }
        __syncthreads();
    }

    // --- final 1/l normalization and store (token, h*64+dh) layout --------
    if (lane < 16) linv_s[lane] = 1.0f / fmaxf(l_st, 1e-30f);
    __syncthreads();
    float li[8];
#pragma unroll
    for (int v = 0; v < 8; ++v) li[v] = linv_s[v + 8 * half];

#pragma unroll
    for (int nt = 0; nt < 4; ++nt)
#pragma unroll
        for (int v = 0; v < 8; ++v) {
            const int m = v + 8 * half;
            O[(long long)(b * N_ + i0 + m) * (H_ * DH_) + h * DH_ + nt * 16 + nn] =
                (bf16_t)(o_acc[nt][v] * li[v]);
        }
}

// ---------------------------------------------------------------------------
// Kernel 6: output projection GEMM  (4096x1024) x (1024x1024) -> f32 out.
// ---------------------------------------------------------------------------
__global__ __launch_bounds__(128) void out_gemm_kernel(
    const bf16_t* __restrict__ O, const bf16_t* __restrict__ WoutT,
    float* __restrict__ out) {
    const int bm   = blockIdx.x & 63;   // 64 row blocks of 64
    const int bn   = blockIdx.x >> 6;   // 16 col blocks of 64
    const int wave = threadIdx.x >> 5;
    const int lane = threadIdx.x & 31;
    const int m0   = bm * 64 + wave * 16;
    const int n0   = bn * 64;

    f32x8 acc[4];
#pragma unroll
    for (int i = 0; i < 4; ++i) acc[i] = f32x8_zero();

    for (int k0 = 0; k0 < D_; k0 += 32) {
        bf16x16 a = frag_a(O, D_, m0, k0);
#pragma unroll
        for (int nt = 0; nt < 4; ++nt) {
            bf16x16 bb = frag_b(WoutT, D_, n0 + nt * 16, k0);
            acc[nt] = wmma_bf16(a, bb, acc[nt]);
        }
    }

    const int half = lane >> 4, nn = lane & 15;
#pragma unroll
    for (int nt = 0; nt < 4; ++nt)
#pragma unroll
        for (int v = 0; v < 8; ++v)
            out[(long long)(m0 + v + 8 * half) * D_ + n0 + nt * 16 + nn] =
                acc[nt][v];
}

// ---------------------------------------------------------------------------
// Host-side launcher
// ---------------------------------------------------------------------------
extern "C" void kernel_launch(void* const* d_in, const int* in_sizes, int n_in,
                              void* d_out, int out_size, void* d_ws, size_t ws_size,
                              hipStream_t stream) {
    const float*         x         = (const float*)d_in[0];
    const float*         attn_bias = (const float*)d_in[1];
    const unsigned char* cmask     = (const unsigned char*)d_in[2]; // bool array
    const float*         gamma     = (const float*)d_in[3];
    const float*         null_kv   = (const float*)d_in[4];
    const float*         Wq        = (const float*)d_in[5];
    const float*         Wkv       = (const float*)d_in[6];
    const float*         q_scale   = (const float*)d_in[7];
    const float*         k_scale   = (const float*)d_in[8];
    const float*         Wout      = (const float*)d_in[9];
    float*               out       = (float*)d_out;
    (void)in_sizes; (void)n_in; (void)out_size; (void)ws_size;

    char*  ws  = (char*)d_ws;
    size_t off = 0;
    auto alloc = [&](size_t bytes) -> void* {
        void* p = ws + off;
        off += (bytes + 255) & ~(size_t)255;
        return p;
    };

    bf16_t* xn    = (bf16_t*)alloc((size_t)B_ * N_ * D_ * 2);
    bf16_t* WqT   = (bf16_t*)alloc((size_t)D_ * (H_ * DH_) * 2);
    bf16_t* WkT   = (bf16_t*)alloc((size_t)D_ * (H_ * DH_) * 2);
    bf16_t* WvT   = (bf16_t*)alloc((size_t)D_ * (H_ * DH_) * 2);
    bf16_t* WoutT = (bf16_t*)alloc((size_t)(H_ * DH_) * D_ * 2);
    bf16_t* Qb    = (bf16_t*)alloc((size_t)B_ * H_ * N_ * DH_ * 2);
    bf16_t* Kb    = (bf16_t*)alloc((size_t)B_ * H_ * LP_ * DH_ * 2);
    bf16_t* Vtb   = (bf16_t*)alloc((size_t)B_ * H_ * DH_ * LP_ * 2);
    bf16_t* Ob    = (bf16_t*)alloc((size_t)B_ * N_ * (H_ * DH_) * 2);

    prep_weights_kernel<<<(D_ * D_) / 256, 256, 0, stream>>>(
        Wq, Wkv, Wout, WqT, WkT, WvT, WoutT);
    ln_kernel<<<B_ * N_, 256, 0, stream>>>(x, gamma, xn);
    init_kv_kernel<<<B_ * H_, 64, 0, stream>>>(null_kv, k_scale, Kb, Vtb);
    qkv_kernel<<<B_ * H_ * (N_ / 16), 96, 0, stream>>>(
        xn, WqT, WkT, WvT, q_scale, k_scale, Qb, Kb, Vtb);
    attn_kernel<<<B_ * H_ * (N_ / 16), 32, 0, stream>>>(
        Qb, Kb, Vtb, attn_bias, cmask, Ob);
    out_gemm_kernel<<<(B_ * N_ / 64) * (D_ / 64), 128, 0, stream>>>(
        Ob, WoutT, out);
}
